// FlowCPAB_23038204576201
// MI455X (gfx1250) — matrix operand order repaired and checked by reading, not compiled, not run on hardware
//
#include <hip/hip_runtime.h>
#include <hip/hip_bf16.h>

#define NCELL 32
#define EPSV  1e-10f

typedef __attribute__((ext_vector_type(16))) _Float16 v16h;
typedef __attribute__((ext_vector_type(8)))  float    v8f;

// ---------------------------------------------------------------------------
// Kernel 1: one wave32. Build the CPA continuity null-space basis (hat
// functions), orthonormalize columns (modified Gram-Schmidt, wave-wide
// __shfl_xor reductions), then compute Atab = B * theta with
// v_wmma_f32_16x16x32_f16 (4 row tiles x 2 K tiles; K padded 33->64 with
// zero-filled columns so fragment packing is branch-free and EXEC stays
// all-ones around the WMMAs).
// ---------------------------------------------------------------------------
__global__ __launch_bounds__(32) void build_A_kernel(const float* __restrict__ theta,
                                                     float* __restrict__ Atab) {
    __shared__ float H[64][64];   // 64 rows (a0,b0,a1,b1,...), cols 33..63 = 0
    __shared__ float th[64];      // theta padded with zeros

    const int L = threadIdx.x;    // lane 0..31

    // Zero-fill, then place the hat basis: column j = CPA field with
    // v(x_k) = delta_{kj}. Cell c (= lane L):
    //   a = NCELL*(d_{c+1,j}-d_{c,j}),  b = d_{c,j} - a*c/NCELL.
    for (int j = 0; j < 64; ++j) { H[2 * L][j] = 0.f; H[2 * L + 1][j] = 0.f; }
    H[2 * L][L]         = -(float)NCELL;
    H[2 * L + 1][L]     = 1.0f + (float)L;
    H[2 * L][L + 1]     =  (float)NCELL;
    H[2 * L + 1][L + 1] = -(float)L;

    th[L]      = (L < 33) ? theta[L] : 0.0f;
    th[L + 32] = (L == 0) ? theta[32] : 0.0f;
    __syncthreads();

    // Modified Gram-Schmidt over 33 columns of length 64 (lane owns rows L, L+32).
    for (int j = 0; j < 33; ++j) {
        for (int i = 0; i < j; ++i) {
            float p = H[L][j] * H[L][i] + H[L + 32][j] * H[L + 32][i];
            for (int off = 16; off > 0; off >>= 1) p += __shfl_xor(p, off, 32);
            H[L][j]      -= p * H[L][i];
            H[L + 32][j] -= p * H[L + 32][i];
        }
        float n2 = H[L][j] * H[L][j] + H[L + 32][j] * H[L + 32][j];
        for (int off = 16; off > 0; off >>= 1) n2 += __shfl_xor(n2, off, 32);
        float inv = rsqrtf(fmaxf(n2, 1e-20f));
        H[L][j]      *= inv;
        H[L + 32][j] *= inv;
    }
    __syncthreads();

    // Atab = H * theta  via v_wmma_f32_16x16x32_f16.
    // A-frag (16x32 f16): lane<16 holds K={0..7,16..23}, lane>=16 K={8..15,24..31}.
    // B-frag (32x16 f16): lane<16 holds K={0..7,16..23} of col (lane&15), lane>=16 K={8..15,24..31}.
    const int   m      = L & 15;
    const bool  hi     = (L >= 16);
    const float colsel = (m == 0) ? 1.0f : 0.0f;   // only column 0 carries theta
    for (int rt = 0; rt < 4; ++rt) {
        v8f acc = {};
        for (int kt = 0; kt < 2; ++kt) {
            v16h af, bf;
            for (int q = 0; q < 16; ++q) {
                const int kk = (q < 8) ? ((hi ? 8 : 0) + q) : ((hi ? 24 : 16) + (q - 8));
                af[q] = (_Float16)H[rt * 16 + m][kt * 32 + kk];        // zero-padded
                bf[q] = (_Float16)(colsel * th[kt * 32 + kk]);
            }
            acc = __builtin_amdgcn_wmma_f32_16x16x32_f16(
                false, af, false, bf, (short)0, acc, false, false);
        }
        // D column 0: lane 0 -> rows rt*16+0..7, lane 16 -> rows rt*16+8..15.
        if (m == 0) {
            const int rbase = rt * 16 + (hi ? 8 : 0);
            for (int q = 0; q < 8; ++q) Atab[rbase + q] = acc[q];
        }
    }
}

// ---------------------------------------------------------------------------
// Kernel 2: per-point closed-form CPAB integration. Transcendental-bound
// (v_exp_f32 / v_log_f32). Wave 0 stages the 256-byte A-table into LDS with
// GLOBAL_LOAD_ASYNC_TO_LDS_B64 (ASYNCcnt path, no VGPR round-trip), the
// block barrier publishes it. NT streaming loads/stores; early-exit once the
// remaining time t reaches 0 (provably a fixed point of the reference scan).
// ---------------------------------------------------------------------------
__global__ __launch_bounds__(256) void cpab_flow_kernel(const float* __restrict__ x_in,
                                                        const float* __restrict__ Atab,
                                                        float* __restrict__ z_out,
                                                        float* __restrict__ lj_out,
                                                        int n) {
    __shared__ float sA[64];      // (a_c, b_c) interleaved
    const int tid = threadIdx.x;

    if (tid < 32) {
        // Low 32 bits of a generic shared pointer are the LDS byte address
        // (ISA 10.2 aperture rule).
        unsigned     lds_addr = (unsigned)(uintptr_t)(&sA[2 * tid]);
        const float* gsrc     = Atab + 2 * tid;
        asm volatile("global_load_async_to_lds_b64 %0, %1, off"
                     :
                     : "v"(lds_addr), "v"(gsrc)
                     : "memory");
        asm volatile("s_wait_asynccnt 0x0" ::: "memory");
    }
    __syncthreads();

    const int i = blockIdx.x * blockDim.x + tid;
    if (i >= n) return;

    const float h = 1.0f / (float)NCELL;
    float xs = __builtin_nontemporal_load(&x_in[i]);
    int c = (int)floorf(xs * (float)NCELL);
    c = min(max(c, 0), NCELL - 1);
    float t  = 1.0f;
    float lj = 0.0f;

    #pragma unroll 1
    for (int s = 0; s < NCELL; ++s) {
        const float2 ab = *(const float2*)&sA[2 * c];     // one ds_load_b64
        const float a = ab.x;
        const float b = ab.y;
        const float v = fmaf(a, xs, b);
        const bool  up  = (v > 0.0f);
        const bool  fin = up ? (c != NCELL - 1) : (c != 0);   // boundary finite?
        const float xc  = up ? (float)(c + 1) * h : (float)c * h;
        const float xcs = fin ? xc : xs;
        const bool  big = (fabsf(a) > EPSV);
        const float a_s = big ? a : 1.0f;
        const float vc  = fmaf(a, xcs, b);
        const float r   = vc / ((v == 0.0f) ? 1.0f : v);
        const float th_a = __logf(fmaxf(r, EPSV)) / a_s;
        const float th_b = (xcs - xs) / ((fabsf(b) > EPSV) ? b : 1.0f);
        float t_hit = big ? ((r > EPSV) ? th_a : __builtin_inff()) : th_b;
        const bool valid = fin && (v != 0.0f) && (t_hit > 0.0f);
        t_hit = valid ? t_hit : __builtin_inff();
        const bool  cross = (t_hit < t);        // cross => t_hit finite (< t <= 1)
        const float dt    = cross ? t_hit : t;
        const float boa   = b / a_s;
        const float psi   = big ? fmaf(__expf(a * dt), xs + boa, -boa)
                                : fmaf(b, dt, xs);
        xs = cross ? xcs : psi;
        lj = fmaf(a, dt, lj);
        t  = cross ? (t - dt) : 0.0f;
        c  = min(max(c + (cross ? (up ? 1 : -1) : 0), 0), NCELL - 1);
        if (t <= 0.0f) break;                   // fixed point of the scan
    }

    __builtin_nontemporal_store(xs, &z_out[i]);
    __builtin_nontemporal_store(lj, &lj_out[i]);
}

extern "C" void kernel_launch(void* const* d_in, const int* in_sizes, int n_in,
                              void* d_out, int out_size, void* d_ws, size_t ws_size,
                              hipStream_t stream) {
    const float* x     = (const float*)d_in[0];   // (N,) f32
    const float* theta = (const float*)d_in[1];   // (1, 33) f32
    float* out  = (float*)d_out;                  // z[0..N) ++ logJ[N..2N)
    float* Atab = (float*)d_ws;                   // 64 floats: (a_c, b_c) interleaved
    const int n = in_sizes[0];

    build_A_kernel<<<1, 32, 0, stream>>>(theta, Atab);

    const int blocks = (n + 255) / 256;
    cpab_flow_kernel<<<blocks, 256, 0, stream>>>(x, Atab, out, out + n, n);
}
